// CausalSelfAttention_26645977105196
// MI455X (gfx1250) — matrix operand order repaired and checked by reading, not compiled, not run on hardware
//
#include <hip/hip_runtime.h>

// ---------------------------------------------------------------------------
// CDNA5 (gfx1250) causal self-attention forward, f16-staged WMMA pipeline.
//   stage 0a: xh   = f16(x)                       [16384,768]
//   stage 0b: WTa  = f16(W_attn^T)                [2304,768]
//   stage 0c: WTp  = f16(W_proj^T)                [768,768]
//   stage 1 : qk(f16) / vT(f16) = x @ W_attn + b  (WMMA, f32 accum)
//             q,k stored [B,T,1536]; v stored transposed [B,H,D,T]
//   stage 2 : yh(f16) = causal_softmax(q k^T / 8) v   (flash, WMMA)
//   stage 3 : out(f32) = yh @ W_proj + b_proj     (WMMA, f32 accum)
// Every WMMA operand is fed by contiguous 16B loads (global_load_b128 /
// ds_load_b128); no per-use f32->f16 conversion in inner loops.
// ---------------------------------------------------------------------------

typedef _Float16 v16h __attribute__((ext_vector_type(16)));
typedef _Float16 v8h  __attribute__((ext_vector_type(8)));
typedef float    v8f  __attribute__((ext_vector_type(8)));
typedef float    v4f  __attribute__((ext_vector_type(4)));

#define C_EMBD   768
#define THREE_C  2304
#define QK_C     1536
#define T_SEQ    1024
#define N_HEAD   12
#define HEAD_D   64
#define B_BATCH  16
#define M_TOK    (B_BATCH * T_SEQ)      // 16384

__device__ __forceinline__ v8f wmma_f16(v16h a, v16h b, v8f c) {
  return __builtin_amdgcn_wmma_f32_16x16x32_f16(false, a, false, b,
                                                (short)0, c, false, false);
}

// 16 contiguous halfs -> v16h (two b128 loads)
__device__ __forceinline__ v16h ld16(const _Float16* p) {
  v8h lo = *(const v8h*)p;
  v8h hi = *(const v8h*)(p + 8);
  return __builtin_shufflevector(lo, hi, 0, 1, 2, 3, 4, 5, 6, 7,
                                 8, 9, 10, 11, 12, 13, 14, 15);
}

// A-operand (16x32 f16, contiguous K order matching B): this lane needs
// K = 2*c0..2*c0+15 (c0 = 8*half) -> two contiguous b128 loads.
__device__ __forceinline__ v16h ld_a(const _Float16* row, int c0) {
  v8h lo = *(const v8h*)(row + 2 * c0);
  v8h hi = *(const v8h*)(row + 2 * c0 + 8);
  return __builtin_shufflevector(lo, hi, 0, 1, 2, 3, 4, 5, 6, 7,
                                 8, 9, 10, 11, 12, 13, 14, 15);
}

// ---------------------------------------------------------------------------
// stage 0a: elementwise f32 -> f16 (vectorized x4)
// ---------------------------------------------------------------------------
__global__ __launch_bounds__(256) void cvt_f16_kernel(
    const float* __restrict__ src, _Float16* __restrict__ dst) {
  const size_t i = ((size_t)blockIdx.x * 256 + threadIdx.x) * 4;
  v4f v = *(const v4f*)(src + i);
  dst[i + 0] = (_Float16)v.x;
  dst[i + 1] = (_Float16)v.y;
  dst[i + 2] = (_Float16)v.z;
  dst[i + 3] = (_Float16)v.w;
}

// ---------------------------------------------------------------------------
// stage 0b/0c: W[K,N] f32 -> WT[N,K] f16.  grid=(N/256, K); coalesced reads.
// ---------------------------------------------------------------------------
__global__ __launch_bounds__(256) void wt_kernel(
    const float* __restrict__ W, _Float16* __restrict__ WT, int K, int N) {
  const int n = blockIdx.x * 256 + threadIdx.x;
  const int k = blockIdx.y;
  WT[(size_t)n * K + k] = (_Float16)W[(size_t)k * N + n];
}

// ---------------------------------------------------------------------------
// stage 1: qkv GEMM.  A = xh[16384,768] f16, B = WTa[2304,768] f16.
// One wave computes a 16x64 strip (A operand reused across 4 WMMAs/K-step).
// cols <  1536 -> qk[B,T,1536] f16 ; cols >= 1536 -> vT[B,H,D,T] f16.
// blockDim=128 (4 waves), grid=(M/16, N/64/4) = (1024, 9).
// ---------------------------------------------------------------------------
__global__ __launch_bounds__(128) void qkv_gemm_kernel(
    const _Float16* __restrict__ A, const _Float16* __restrict__ WT,
    const float* __restrict__ bias, _Float16* __restrict__ qk,
    _Float16* __restrict__ vT) {
  const int lane = threadIdx.x & 31;
  const int wv   = threadIdx.x >> 5;
  const int half = lane >> 4;
  const int ln   = lane & 15;
  const int c0   = half * 8;

  const int mt   = blockIdx.x;
  const int nt64 = blockIdx.y * 4 + wv;          // 64-col strip index

  const _Float16* arow = A + (size_t)(mt * 16 + ln) * C_EMBD;
  // B rows (= output columns) handled by this lane
  const _Float16* brow[4];
#pragma unroll
  for (int j = 0; j < 4; ++j)
    brow[j] = WT + (size_t)(nt64 * 64 + j * 16 + ln) * C_EMBD + 16 * half;

  v8f acc[4] = {v8f{}, v8f{}, v8f{}, v8f{}};
  for (int k0 = 0; k0 < C_EMBD; k0 += 32) {
    if (k0 + 32 < C_EMBD) {                      // gfx1250 global_prefetch_b8
      __builtin_prefetch(arow + k0 + 32, 0, 3);
      __builtin_prefetch(brow[0] + k0 + 32, 0, 3);
    }
    const v16h av = ld_a(arow + k0, c0);
#pragma unroll
    for (int j = 0; j < 4; ++j)
      acc[j] = wmma_f16(av, ld16(brow[j] + k0), acc[j]);
  }

  const int row0 = mt * 16 + 8 * half;           // first output row (token)
#pragma unroll
  for (int j = 0; j < 4; ++j) {
    const int col = nt64 * 64 + j * 16 + ln;
    const float bv = bias[col];
    if (col < QK_C) {                            // q or k -> [B,T,1536]
#pragma unroll
      for (int r = 0; r < 8; ++r)
        qk[(size_t)(row0 + r) * QK_C + col] = (_Float16)(acc[j][r] + bv);
    } else {                                     // v -> transposed [B,H,D,T]
      const int dall = col - QK_C;               // h*64 + d
      const int b = row0 >> 10, t0 = row0 & 1023;
      v8h pack;
#pragma unroll
      for (int r = 0; r < 8; ++r) pack[r] = (_Float16)(acc[j][r] + bv);
      *(v8h*)(vT + ((size_t)b * (N_HEAD * HEAD_D) + dall) * T_SEQ + t0) = pack;
    }
  }
}

// ---------------------------------------------------------------------------
// stage 2: flash-style causal attention. One wave per 16-query tile.
// qk: [B,T,1536] f16 (q cols 0..767, k cols 768..1535), vT: [B,H,D,T] f16.
// grid = (B*H*(T/16)/8) blocks of 256 threads. Writes yh [B,T,C] f16.
// ---------------------------------------------------------------------------
__global__ __launch_bounds__(256) void attn_kernel(
    const _Float16* __restrict__ qk, const _Float16* __restrict__ vT,
    _Float16* __restrict__ yh) {
  __shared__ _Float16 pscratch[8][16 * 32];      // 1 KB P-tile per wave

  const int lane = threadIdx.x & 31;
  const int wv   = threadIdx.x >> 5;
  const int half = lane >> 4;
  const int ln   = lane & 15;
  const int c0   = half * 8;

  const int wid   = blockIdx.x * 8 + wv;
  const int qt    = wid & 63;
  const int h     = (wid >> 6) % N_HEAD;
  const int b     = wid / (64 * N_HEAD);
  const int qbase = qt * 16;

  const _Float16* qkb  = qk + (size_t)b * T_SEQ * QK_C;
  const _Float16* qrow = qkb + (size_t)(qbase + ln) * QK_C + h * HEAD_D;
  const v16h qa0 = ld_a(qrow, c0);               // d 0..31
  const v16h qa1 = ld_a(qrow + 32, c0);          // d 32..63

  // V rows for this lane's 4 output d-columns: vT[b][h][dt*16+ln][*]
  const _Float16* vrow[4];
#pragma unroll
  for (int dt = 0; dt < 4; ++dt)
    vrow[dt] = vT + ((size_t)(b * N_HEAD + h) * HEAD_D + dt * 16 + ln) * T_SEQ
             + 16 * half;

  v8f o[4] = {v8f{}, v8f{}, v8f{}, v8f{}};
  float m[8], lsum[8];
#pragma unroll
  for (int r = 0; r < 8; ++r) { m[r] = -3.0e38f; lsum[r] = 0.0f; }

  _Float16* pbuf = pscratch[wv];
  const int nsteps = qbase / 32 + 1;

  for (int s = 0; s < nsteps; ++s) {
    const int kb = s * 32;

    // ---- S = Q K^T (32 keys): B-operand = key rows, contiguous in d
    const _Float16* krow0 = qkb + (size_t)(kb + ln) * QK_C + C_EMBD
                          + h * HEAD_D + 16 * half;
    const _Float16* krow1 = krow0 + (size_t)16 * QK_C;
    v8f s0 = {}, s1 = {};
    s0 = wmma_f16(qa0, ld16(krow0), s0);
    s0 = wmma_f16(qa1, ld16(krow0 + 32), s0);
    s1 = wmma_f16(qa0, ld16(krow1), s1);
    s1 = wmma_f16(qa1, ld16(krow1 + 32), s1);

    // ---- scale + causal mask + row max (rows live in one 16-lane half)
    float rowmax[8];
#pragma unroll
    for (int r = 0; r < 8; ++r) {
      const int q = qbase + 8 * half + r;
      const float a0 = (kb + ln      <= q) ? s0[r] * 0.125f : -3.0e38f;
      const float a1 = (kb + 16 + ln <= q) ? s1[r] * 0.125f : -3.0e38f;
      s0[r] = a0; s1[r] = a1;
      rowmax[r] = fmaxf(a0, a1);
    }
#pragma unroll
    for (int mk = 1; mk < 16; mk <<= 1)
#pragma unroll
      for (int r = 0; r < 8; ++r)
        rowmax[r] = fmaxf(rowmax[r], __shfl_xor(rowmax[r], mk, 32));

    // ---- online softmax update
    float alpha[8], rsum[8];
#pragma unroll
    for (int r = 0; r < 8; ++r) {
      const float mn = fmaxf(m[r], rowmax[r]);
      alpha[r] = __expf(m[r] - mn);
      m[r] = mn;
      const float p0 = __expf(s0[r] - mn);
      const float p1 = __expf(s1[r] - mn);
      s0[r] = p0; s1[r] = p1;
      rsum[r] = p0 + p1;
    }
#pragma unroll
    for (int mk = 1; mk < 16; mk <<= 1)
#pragma unroll
      for (int r = 0; r < 8; ++r)
        rsum[r] += __shfl_xor(rsum[r], mk, 32);
#pragma unroll
    for (int r = 0; r < 8; ++r) {
      lsum[r] = lsum[r] * alpha[r] + rsum[r];
      o[0][r] *= alpha[r]; o[1][r] *= alpha[r];
      o[2][r] *= alpha[r]; o[3][r] *= alpha[r];
    }

    // ---- C/D-layout P -> LDS -> A-layout P (same-wave DS ops in order)
#pragma unroll
    for (int r = 0; r < 8; ++r) {
      const int row = 8 * half + r;
      pbuf[row * 32 + ln]      = (_Float16)s0[r];
      pbuf[row * 32 + 16 + ln] = (_Float16)s1[r];
    }
    __builtin_amdgcn_wave_barrier();
    const v16h pa = ld_a(pbuf + ln * 32, c0);    // 2x ds_load_b128
    __builtin_amdgcn_wave_barrier();

    // ---- O += P V : B-operand = vT rows, contiguous in key index
#pragma unroll
    for (int dt = 0; dt < 4; ++dt)
      o[dt] = wmma_f16(pa, ld16(vrow[dt] + kb), o[dt]);
  }

  // ---- normalize, write yh [B,T,C] f16
  _Float16* yrow = yh + ((size_t)b * T_SEQ + qbase + 8 * half) * C_EMBD
                 + h * HEAD_D;
#pragma unroll
  for (int r = 0; r < 8; ++r) {
    const float inv = 1.0f / lsum[r];
#pragma unroll
    for (int dt = 0; dt < 4; ++dt)
      yrow[(size_t)r * C_EMBD + dt * 16 + ln] = (_Float16)(o[dt][r] * inv);
  }
}

// ---------------------------------------------------------------------------
// stage 3: out = yh @ W_proj + b_proj.  A = yh f16, B = WTp[768,768] f16.
// One wave per 16x64 strip. blockDim=128, grid=(1024, 3). f32 output.
// ---------------------------------------------------------------------------
__global__ __launch_bounds__(128) void proj_gemm_kernel(
    const _Float16* __restrict__ A, const _Float16* __restrict__ WT,
    const float* __restrict__ bias, float* __restrict__ out) {
  const int lane = threadIdx.x & 31;
  const int wv   = threadIdx.x >> 5;
  const int half = lane >> 4;
  const int ln   = lane & 15;
  const int c0   = half * 8;

  const int mt   = blockIdx.x;
  const int nt64 = blockIdx.y * 4 + wv;

  const _Float16* arow = A + (size_t)(mt * 16 + ln) * C_EMBD;
  const _Float16* brow[4];
#pragma unroll
  for (int j = 0; j < 4; ++j)
    brow[j] = WT + (size_t)(nt64 * 64 + j * 16 + ln) * C_EMBD + 16 * half;

  v8f acc[4] = {v8f{}, v8f{}, v8f{}, v8f{}};
  for (int k0 = 0; k0 < C_EMBD; k0 += 32) {
    if (k0 + 32 < C_EMBD) {
      __builtin_prefetch(arow + k0 + 32, 0, 3);
      __builtin_prefetch(brow[0] + k0 + 32, 0, 3);
    }
    const v16h av = ld_a(arow + k0, c0);
#pragma unroll
    for (int j = 0; j < 4; ++j)
      acc[j] = wmma_f16(av, ld16(brow[j] + k0), acc[j]);
  }

  const int row0 = mt * 16 + 8 * half;
#pragma unroll
  for (int j = 0; j < 4; ++j) {
    const int col = nt64 * 64 + j * 16 + ln;
    const float bv = bias[col];
#pragma unroll
    for (int r = 0; r < 8; ++r)
      out[(size_t)(row0 + r) * C_EMBD + col] = acc[j][r] + bv;
  }
}

// ---------------------------------------------------------------------------
extern "C" void kernel_launch(void* const* d_in, const int* in_sizes, int n_in,
                              void* d_out, int out_size, void* d_ws,
                              size_t ws_size, hipStream_t stream) {
  const float* x      = (const float*)d_in[0];   // [16,1024,768]
  const float* W_attn = (const float*)d_in[1];   // [768,2304]
  const float* b_attn = (const float*)d_in[2];   // [2304]
  const float* W_proj = (const float*)d_in[3];   // [768,768]
  const float* b_proj = (const float*)d_in[4];   // [768]
  float* out = (float*)d_out;                    // [16,1024,768]

  // workspace layout (halfs); all offsets 16B-aligned
  _Float16* xh  = (_Float16*)d_ws;                               // 12.58M
  _Float16* WTa = xh  + (size_t)M_TOK * C_EMBD;                  //  1.77M
  _Float16* WTp = WTa + (size_t)THREE_C * C_EMBD;                //  0.59M
  _Float16* qkb = WTp + (size_t)C_EMBD * C_EMBD;                 // 25.17M
  _Float16* vT  = qkb + (size_t)M_TOK * QK_C;                    // 12.58M
  _Float16* yh  = vT  + (size_t)M_TOK * HEAD_D * N_HEAD / 64 * 64; // 12.58M

  // stage 0: f16 staging
  cvt_f16_kernel<<<(M_TOK * C_EMBD) / 4 / 256, 256, 0, stream>>>(x, xh);
  wt_kernel<<<dim3(THREE_C / 256, C_EMBD), 256, 0, stream>>>(W_attn, WTa,
                                                             C_EMBD, THREE_C);
  wt_kernel<<<dim3(C_EMBD / 256, C_EMBD), 256, 0, stream>>>(W_proj, WTp,
                                                            C_EMBD, C_EMBD);

  // stage 1: qkv GEMM -> qk (f16) + transposed v (f16)
  qkv_gemm_kernel<<<dim3(M_TOK / 16, THREE_C / 64 / 4), 128, 0, stream>>>(
      xh, WTa, b_attn, qkb, vT);

  // stage 2: attention (B*H*T/16 = 12288 waves -> 1536 blocks of 8 waves)
  attn_kernel<<<(B_BATCH * N_HEAD * (T_SEQ / 16)) / 8, 256, 0, stream>>>(
      qkb, vT, yh);

  // stage 3: out = yh @ W_proj + b_proj
  proj_gemm_kernel<<<dim3(M_TOK / 16, C_EMBD / 64 / 4), 128, 0, stream>>>(
      yh, WTp, b_proj, out);
}
